// KappaGCNLayer_24653112279102
// MI455X (gfx1250) — matrix operand order repaired and checked by reading, not compiled, not run on hardware
//
#include <hip/hip_runtime.h>

typedef __attribute__((ext_vector_type(16))) __bf16 v16bf;
typedef __attribute__((ext_vector_type(8)))  float  v8f;
typedef __attribute__((ext_vector_type(4)))  float  f32x4;

#define NPTS  16384
#define DIN   64
#define DOUT  32
#define MTC   48          // padded GEMM output columns: 32 nom + den + alpha + 14 pad
#define EPSN  1e-15f
#define BEPS  1e-7f

__device__ __forceinline__ float artanh_c(float x) {
    x = fminf(fmaxf(x, -1.0f + BEPS), 1.0f - BEPS);
    return 0.5f * logf((1.0f + x) / (1.0f - x));
}

// ---------------------------------------------------------------------------
// Kernel 1: mobius_matvec(W, X) -> hyperbolic points h, conformal factor gamma.
// Writes transposed, split-bf16 matrix Mt (96 rows x 16384):
//   rows 0..31  : hi(gamma_j * h_j[c])      rows 48..79 : lo residual
//   row  32     : hi(gamma_j - 1)           row  80     : lo residual
//   row  33     : 1.0                       row  81     : 0
//   rows 34..47 : 0                         rows 82..95 : 0
// ---------------------------------------------------------------------------
__global__ void __launch_bounds__(256)
k1_features(const float* __restrict__ X,
            const float* __restrict__ W,
            __bf16* __restrict__ Mt)
{
    __shared__ float sW[DIN * DOUT];
    for (int i = threadIdx.x; i < DIN * DOUT; i += 256) sW[i] = W[i];
    __syncthreads();

    const int j = blockIdx.x * 256 + threadIdx.x;
    const float* x = X + (size_t)j * DIN;

    float mx[DOUT];
    #pragma unroll
    for (int c = 0; c < DOUT; ++c) mx[c] = 0.0f;
    float xn2 = 0.0f;
    for (int k = 0; k < DIN; ++k) {
        float xk = x[k];
        xn2 = fmaf(xk, xk, xn2);
        #pragma unroll
        for (int c = 0; c < DOUT; ++c) mx[c] = fmaf(xk, sW[k * DOUT + c], mx[c]);
    }
    float xn = fmaxf(sqrtf(xn2), EPSN);
    float mxn2 = 0.0f;
    #pragma unroll
    for (int c = 0; c < DOUT; ++c) mxn2 = fmaf(mx[c], mx[c], mxn2);
    float mxn = fmaxf(sqrtf(mxn2), EPSN);
    float scale = tanhf(mxn / xn * artanh_c(xn)) / mxn;

    float h[DOUT]; float h2 = 0.0f;
    #pragma unroll
    for (int c = 0; c < DOUT; ++c) { h[c] = scale * mx[c]; h2 = fmaf(h[c], h[c], h2); }
    float gamma = 2.0f / fmaxf(1.0f - h2, EPSN);   // k = -1 conformal factor

    #pragma unroll
    for (int c = 0; c < DOUT; ++c) {
        float vv = gamma * h[c];
        __bf16 hi = (__bf16)vv;
        Mt[(size_t)c * NPTS + j]        = hi;
        Mt[(size_t)(c + 48) * NPTS + j] = (__bf16)(vv - (float)hi);
    }
    {
        float vv = gamma - 1.0f;
        __bf16 hi = (__bf16)vv;
        Mt[(size_t)32 * NPTS + j]        = hi;
        Mt[(size_t)(32 + 48) * NPTS + j] = (__bf16)(vv - (float)hi);
    }
    Mt[(size_t)33 * NPTS + j]        = (__bf16)1.0f;   // alpha = A @ 1 column
    Mt[(size_t)(33 + 48) * NPTS + j] = (__bf16)0.0f;
    #pragma unroll
    for (int c = 34; c < 48; ++c) {
        Mt[(size_t)c * NPTS + j]        = (__bf16)0.0f;
        Mt[(size_t)(c + 48) * NPTS + j] = (__bf16)0.0f;
    }
}

// ---------------------------------------------------------------------------
// Kernel 2: Oraw(16384 x 48) = A(16384 x 16384, fp32, streamed NT) x M(16384 x 48)
// Split-bf16 WMMA: A=ah+al, M=bh+bl, acc += ah*bh + al*bh + ah*bl.
// Each wave: 32 rows x 48 cols; 8 waves/block -> 256 rows; 64 blocks.
// ---------------------------------------------------------------------------
__global__ void __launch_bounds__(256)
k2_gemm_wmma(const float* __restrict__ A,
             const __bf16* __restrict__ Mt,
             float* __restrict__ Oraw)
{
    const int lane = threadIdx.x & 31;
    const int wave = threadIdx.x >> 5;
    const int r0   = blockIdx.x * 256 + wave * 32;

    const int am  = lane & 15;           // A-fragment row within tile
    const int akb = (lane >> 4) * 16;    // A-fragment K sub-base (0 or 16)
    const int bn  = lane & 15;           // B-fragment column within tile
    const int bkb = (lane >> 4) * 16;    // B-fragment K sub-base (0 or 16)

    v8f acc[2][3] = {};                  // 2 row-tiles x 3 col-tiles

    const float* arow0 = A + (size_t)(r0 + am) * NPTS;
    const float* arow1 = arow0 + (size_t)16 * NPTS;

    for (int k0 = 0; k0 < NPTS; k0 += 32) {
        // --- A fragments: fp32 stream (non-temporal), split into bf16 hi+lo ---
        v16bf ah[2], al[2];
        #pragma unroll
        for (int f = 0; f < 2; ++f) {
            const float* ar = (f == 0) ? arow0 : arow1;
            f32x4 q0 = __builtin_nontemporal_load((const f32x4*)(ar + k0 + akb));
            f32x4 q1 = __builtin_nontemporal_load((const f32x4*)(ar + k0 + akb) + 1);
            f32x4 q2 = __builtin_nontemporal_load((const f32x4*)(ar + k0 + akb) + 2);
            f32x4 q3 = __builtin_nontemporal_load((const f32x4*)(ar + k0 + akb) + 3);
            float v[16] = {q0[0],q0[1],q0[2],q0[3], q1[0],q1[1],q1[2],q1[3],
                           q2[0],q2[1],q2[2],q2[3], q3[0],q3[1],q3[2],q3[3]};
            #pragma unroll
            for (int i = 0; i < 16; ++i) {
                __bf16 hi = (__bf16)v[i];
                ah[f][i] = hi;
                al[f][i] = (__bf16)(v[i] - (float)hi);
            }
        }

        // --- B fragments from L2-resident transposed M (32B contiguous/lane) ---
        #pragma unroll
        for (int t = 0; t < 3; ++t) {
            const __bf16* bp = Mt + (size_t)(t * 16 + bn) * NPTS + k0 + bkb;
            v16bf bh = *(const v16bf*)bp;
            v16bf bl = *(const v16bf*)(bp + (size_t)48 * NPTS);
            #pragma unroll
            for (int f = 0; f < 2; ++f) {
                acc[f][t] = __builtin_amdgcn_wmma_f32_16x16x32_bf16(
                    false, ah[f], false, bh, (short)0, acc[f][t], false, false);
                acc[f][t] = __builtin_amdgcn_wmma_f32_16x16x32_bf16(
                    false, al[f], false, bh, (short)0, acc[f][t], false, false);
                acc[f][t] = __builtin_amdgcn_wmma_f32_16x16x32_bf16(
                    false, ah[f], false, bl, (short)0, acc[f][t], false, false);
            }
        }
    }

    // D layout: VGPR r -> M = base + (lane>>4)*8 + r, N = lane&15
    const int mb = (lane >> 4) * 8;
    #pragma unroll
    for (int f = 0; f < 2; ++f) {
        #pragma unroll
        for (int r = 0; r < 8; ++r) {
            float* orow = Oraw + (size_t)(r0 + f * 16 + mb + r) * MTC;
            #pragma unroll
            for (int t = 0; t < 3; ++t)
                orow[t * 16 + bn] = acc[f][t][r];
        }
    }
}

// ---------------------------------------------------------------------------
// Kernel 3: weighted-midpoint epilogue + mobius scalar muls + manifold relu.
// ---------------------------------------------------------------------------
__global__ void __launch_bounds__(256)
k3_midpoint_relu(const float* __restrict__ Oraw, float* __restrict__ out)
{
    const int i = blockIdx.x * 256 + threadIdx.x;
    const float* row = Oraw + (size_t)i * MTC;

    float den   = row[32];
    float alpha = row[33];
    float sgn = (den >= 0.0f) ? 1.0f : -1.0f;
    den = sgn * fmaxf(fabsf(den), 1e-10f);

    float two[DOUT]; float tn2 = 0.0f;
    #pragma unroll
    for (int c = 0; c < DOUT; ++c) { two[c] = row[c] / den; tn2 = fmaf(two[c], two[c], tn2); }
    float tn = fmaxf(sqrtf(tn2), EPSN);
    float s1 = tanhf(0.5f * artanh_c(tn)) / tn;            // mobius_scalar_mul(0.5, .)

    float am2[DOUT]; float an2 = 0.0f;
    #pragma unroll
    for (int c = 0; c < DOUT; ++c) { am2[c] = s1 * two[c]; an2 = fmaf(am2[c], am2[c], an2); }
    float an = fmaxf(sqrtf(an2), EPSN);
    float s2 = tanhf(alpha * artanh_c(an)) / an;           // mobius_scalar_mul(alpha, .)

    float ym[DOUT]; float yn2 = 0.0f;
    #pragma unroll
    for (int c = 0; c < DOUT; ++c) { ym[c] = s2 * am2[c]; yn2 = fmaf(ym[c], ym[c], yn2); }
    float yn = fmaxf(sqrtf(yn2), EPSN);
    float su = artanh_c(yn) / yn;                          // logmap0

    float u[DOUT]; float un2 = 0.0f;
    #pragma unroll
    for (int c = 0; c < DOUT; ++c) { u[c] = fmaxf(su * ym[c], 0.0f); un2 = fmaf(u[c], u[c], un2); }
    float un = fmaxf(sqrtf(un2), EPSN);
    float so = tanhf(un) / un;                             // expmap0

    float* o = out + (size_t)i * DOUT;
    #pragma unroll
    for (int c = 0; c < DOUT; ++c) o[c] = so * u[c];
}

// ---------------------------------------------------------------------------
extern "C" void kernel_launch(void* const* d_in, const int* in_sizes, int n_in,
                              void* d_out, int out_size, void* d_ws, size_t ws_size,
                              hipStream_t stream)
{
    const float* X = (const float*)d_in[0];     // (16384, 64)
    const float* A = (const float*)d_in[1];     // (16384, 16384)
    const float* W = (const float*)d_in[2];     // (64, 32)
    float* out = (float*)d_out;                 // (16384, 32)

    __bf16* Mt = (__bf16*)d_ws;                              // 96 x 16384 bf16 = 3 MB
    size_t mt_bytes = ((size_t)96 * NPTS * sizeof(__bf16) + 255) & ~(size_t)255;
    float* Oraw = (float*)((char*)d_ws + mt_bytes);          // 16384 x 48 f32 = 3 MB

    hipLaunchKernelGGL(k1_features,      dim3(NPTS / 256), dim3(256), 0, stream, X, W, Mt);
    hipLaunchKernelGGL(k2_gemm_wmma,     dim3(NPTS / 256), dim3(256), 0, stream, A, Mt, Oraw);
    hipLaunchKernelGGL(k3_midpoint_relu, dim3(NPTS / 256), dim3(256), 0, stream, Oraw, out);
}
